// simpleSNN_78692390797944
// MI455X (gfx1250) — matrix operand order repaired and checked by reading, not compiled, not run on hardware
//
#include <hip/hip_runtime.h>
#include <hip/hip_bf16.h>
#include <float.h>

typedef __attribute__((ext_vector_type(16))) _Float16 v16h;
typedef __attribute__((ext_vector_type(8)))  float    v8f;

#define T_STEPS 100
#define NWG     13      // 12 layer-1 WGs (96 waves = 96 N-tiles) + 1 layer-2/3 WG
#define NK1     96      // K-tiles layer1  (3072/32)
#define NK2     48      // K-tiles layer2  (1536/32)
#define NK3     8       // K-tiles layer3  (256/32)
#define NT1     96      // N-tiles layer1  (1536/16)
#define NT2     16      // N-tiles layer2  (256/16)

// ---- workspace layout (bytes), all 64B aligned ----
static constexpr size_t OFF_BAR    = 0;
static constexpr size_t OFF_Z1A    = 64;
static constexpr size_t Z1A_HALVES = 2ull * NK2 * 32 * 16;           // double-buffered z1 A-fragments
static constexpr size_t OFF_W1H    = OFF_Z1A + Z1A_HALVES * 2;
static constexpr size_t W1H_HALVES = (size_t)NT1 * NK1 * 32 * 16;    // 9.44 MB
static constexpr size_t OFF_W2H    = OFF_W1H + W1H_HALVES * 2;
static constexpr size_t W2H_HALVES = (size_t)NT2 * NK2 * 32 * 16;
static constexpr size_t OFF_W3H    = OFF_W2H + W2H_HALVES * 2;
static constexpr size_t W3H_HALVES = (size_t)1 * NK3 * 32 * 16;
static constexpr size_t OFF_SPK    = OFF_W3H + W3H_HALVES * 2;
static constexpr size_t SPK_HALVES = (size_t)T_STEPS * NK1 * 32 * 16; // 9.83 MB
static constexpr size_t WS_NEEDED  = OFF_SPK + SPK_HALVES * 2;

// A-matrix (16x32 f16) per-lane K mapping from the CDNA5 ISA table:
// lanes 0-15: V0..3 -> K 0..7, V4..7 -> K 16..23; lanes 16-31: +8.
__device__ __forceinline__ int koffA(int lane, int i) {
  return i + (i & 8) + ((lane & 16) >> 1);
}
// B-matrix (32x16 f16): same K mapping as A: lanes 0-15: i 0..7 -> K 0..7, i 8..15 -> K 16..23; lanes 16-31: +8.
__device__ __forceinline__ int koffB(int lane, int i) {
  return i + (i & 8) + ((lane & 16) >> 1);
}

// ---------- one-time prep kernels ----------
__global__ void snn_init(char* __restrict__ ws) {
  size_t idx = (size_t)blockIdx.x * blockDim.x + threadIdx.x;
  if (idx == 0) *(unsigned int*)(ws + OFF_BAR) = 0u;
  _Float16* z1A = (_Float16*)(ws + OFF_Z1A);
  if (idx < Z1A_HALVES) z1A[idx] = (_Float16)0.f;   // pad rows M=4..15 stay zero
}

// pack W (out_d x in_d, row-major f32) into f16 B-fragment order
__global__ void pack_w(const float* __restrict__ W, _Float16* __restrict__ Wh,
                       int nKt, int in_d, int out_d, long total) {
  long idx = (long)blockIdx.x * blockDim.x + threadIdx.x;
  if (idx >= total) return;
  int  i    = (int)(idx & 15);
  int  lane = (int)((idx >> 4) & 31);
  long r    = idx >> 9;
  int  kt   = (int)(r % nKt);
  int  nt   = (int)(r / nKt);
  int  o    = nt * 16 + (lane & 15);
  int  j    = kt * 32 + koffB(lane, i);
  float val = (o < out_d) ? W[(size_t)o * in_d + j] : 0.f;   // pad N for layer3 (10->16)
  Wh[idx] = (_Float16)val;
}

// Poisson-encode all T steps directly into A-fragment order (rows = batch, M=4..15 zero)
__global__ void encode_spikes(const float* __restrict__ u, const float* __restrict__ x,
                              _Float16* __restrict__ spk) {
  long idx = (long)blockIdx.x * blockDim.x + threadIdx.x;
  if (idx >= (long)SPK_HALVES) return;
  int  i    = (int)(idx & 15);
  int  lane = (int)((idx >> 4) & 31);
  long r    = idx >> 9;
  int  kt   = (int)(r % NK1);
  int  t    = (int)(r / NK1);
  int  b    = lane & 15;
  _Float16 s = (_Float16)0.f;
  if (b < 4) {
    int j = kt * 32 + koffA(lane, i);
    float thr = x[(size_t)b * 3072 + j] * 0.02f;             // F_MAX * DT
    s = (u[((size_t)t * 4 + b) * 3072 + j] < thr) ? (_Float16)1.f : (_Float16)0.f;
  }
  spk[idx] = s;
}

// ---------- persistent main kernel ----------
__device__ __forceinline__ void grid_barrier(unsigned int* bar, unsigned int target) {
  __threadfence();
  __syncthreads();
  if (threadIdx.x == 0) {
    __hip_atomic_fetch_add(bar, 1u, __ATOMIC_ACQ_REL, __HIP_MEMORY_SCOPE_AGENT);
    while (__hip_atomic_load(bar, __ATOMIC_ACQUIRE, __HIP_MEMORY_SCOPE_AGENT) < target) {
      __builtin_amdgcn_s_sleep(2);
    }
  }
  __syncthreads();
  __threadfence();
}

__device__ __forceinline__ v8f wmma16(v16h a, v16h b, v8f c) {
  return __builtin_amdgcn_wmma_f32_16x16x32_f16(false, a, false, b, (short)0, c, false, false);
}

__global__ __launch_bounds__(256, 1)
void snn_main(const float* __restrict__ b1, const float* __restrict__ b2,
              const float* __restrict__ b3, float* __restrict__ out,
              char* __restrict__ ws) {
  unsigned int* bar = (unsigned int*)(ws + OFF_BAR);
  _Float16*     z1A = (_Float16*)(ws + OFF_Z1A);
  const int tid  = threadIdx.x;
  const int lane = tid & 31;
  const int wave = tid >> 5;
  const int wg   = blockIdx.x;

  __shared__ __align__(32) _Float16 z2A[NK3 * 32 * 16];   // z2 A-fragments, 8 KB LDS
  if (wg == NWG - 1) {
    for (int k = tid; k < NK3 * 32 * 16; k += 256) z2A[k] = (_Float16)0.f;
  }
  __syncthreads();

  if (wg < 12) {
    // ---------------- layer 1: one 16-wide N-tile per wave, LIF state in regs ----------------
    const int nt = wg * 8 + wave;
    const int n0 = nt * 16;
    const float bias = b1[n0 + (lane & 15)];
    float v[4]  = {0.f, 0.f, 0.f, 0.f};
    float cc[4] = {0.f, 0.f, 0.f, 0.f};
    const v16h* Wb = (const v16h*)(ws + OFF_W1H) + (size_t)nt * NK1 * 32 + lane;
    const v16h* Sb = (const v16h*)(ws + OFF_SPK) + lane;

    // precompute scatter slot for writing this lane's z1 into A-fragment order
    const int n   = n0 + (lane & 15);
    const int kt2 = n >> 5;
    const int k5  = n & 31;
    const int g   = (k5 >> 3) & 1;                 // inverse of koffA
    const int ii  = (k5 & 7) | ((k5 & 16) >> 1);

    for (int t = 0; t < T_STEPS; ++t) {
      v8f acc = {0.f, 0.f, 0.f, 0.f, 0.f, 0.f, 0.f, 0.f};
      const v16h* Ap = Sb + (size_t)t * NK1 * 32;
      const v16h* Bp = Wb;
#pragma unroll 8
      for (int kt = 0; kt < NK1; ++kt) {
        __builtin_prefetch((const void*)(Bp + 32 * 8), 0, 1);   // global_prefetch_b8
        v16h a = Ap[0];
        v16h b = Bp[0];
        acc = wmma16(a, b, acc);
        Ap += 32; Bp += 32;
      }
      float z[4];
#pragma unroll
      for (int bb = 0; bb < 4; ++bb) {            // LIF: decay, spike, reset, inject
        float vd = v[bb] + 0.1f * (cc[bb] - v[bb]);
        float id = 0.8f * cc[bb];
        float zz = ((vd - 0.4f) > 0.f) ? 1.f : 0.f;
        v[bb]  = (1.f - zz) * vd;
        cc[bb] = id + acc[bb] + bias;
        z[bb]  = zz;
      }
      if (lane < 16) {
        _Float16* dst = z1A + ((size_t)((t & 1) * NK2 + kt2) * 32) * 16;
#pragma unroll
        for (int bb = 0; bb < 4; ++bb)
          dst[(bb + (g << 4)) * 16 + ii] = (_Float16)z[bb];
      }
      grid_barrier(bar, (unsigned)(NWG * (t + 1)));
    }
  } else {
    // ---------------- WG 12: layer 2 (8 waves x 2 tiles) + layer 3 (wave 0), one step behind ----
    float v2[2][4], c2[2][4], bias2[2];
#pragma unroll
    for (int s = 0; s < 2; ++s) {
      bias2[s] = b2[(wave * 2 + s) * 16 + (lane & 15)];
#pragma unroll
      for (int bb = 0; bb < 4; ++bb) { v2[s][bb] = 0.f; c2[s][bb] = 0.f; }
    }
    float v3[4]   = {0.f, 0.f, 0.f, 0.f};
    float c3[4]   = {0.f, 0.f, 0.f, 0.f};
    float vmax[4] = {-FLT_MAX, -FLT_MAX, -FLT_MAX, -FLT_MAX};
    const float bias3 = ((lane & 15) < 10) ? b3[lane & 15] : 0.f;

    for (int t = 0; t <= T_STEPS; ++t) {
      if (t >= 1) {
        const int buf = (t - 1) & 1;
#pragma unroll
        for (int s = 0; s < 2; ++s) {
          const int nt2 = wave * 2 + s;
          v8f acc = {0.f, 0.f, 0.f, 0.f, 0.f, 0.f, 0.f, 0.f};
          const v16h* Ap = (const v16h*)z1A + (size_t)buf * NK2 * 32 + lane;
          const v16h* Bp = (const v16h*)(ws + OFF_W2H) + (size_t)nt2 * NK2 * 32 + lane;
#pragma unroll 4
          for (int kt = 0; kt < NK2; ++kt) {
            acc = wmma16(Ap[0], Bp[0], acc);
            Ap += 32; Bp += 32;
          }
          float z[4];
#pragma unroll
          for (int bb = 0; bb < 4; ++bb) {
            float vd = v2[s][bb] + 0.1f * (c2[s][bb] - v2[s][bb]);
            float id = 0.8f * c2[s][bb];
            float zz = ((vd - 0.4f) > 0.f) ? 1.f : 0.f;
            v2[s][bb] = (1.f - zz) * vd;
            c2[s][bb] = id + acc[bb] + bias2[s];
            z[bb] = zz;
          }
          if (lane < 16) {
            int n2  = nt2 * 16 + lane;
            int kt3 = n2 >> 5;
            int k5  = n2 & 31;
            int gg  = (k5 >> 3) & 1;
            int jj  = (k5 & 7) | ((k5 & 16) >> 1);
#pragma unroll
            for (int bb = 0; bb < 4; ++bb)
              z2A[((kt3 * 32) + bb + (gg << 4)) * 16 + jj] = (_Float16)z[bb];
          }
        }
        __syncthreads();
        if (wave == 0) {                           // layer 3: LI readout, running max
          v8f acc = {0.f, 0.f, 0.f, 0.f, 0.f, 0.f, 0.f, 0.f};
          const v16h* Ap = (const v16h*)z2A + lane;
          const v16h* Bp = (const v16h*)(ws + OFF_W3H) + lane;
#pragma unroll
          for (int kt = 0; kt < NK3; ++kt) {
            acc = wmma16(Ap[0], Bp[0], acc);
            Ap += 32; Bp += 32;
          }
#pragma unroll
          for (int bb = 0; bb < 4; ++bb) {
            float vn = v3[bb] + 0.1f * (c3[bb] - v3[bb]);
            c3[bb] = 0.8f * c3[bb] + acc[bb] + bias3;
            v3[bb] = vn;
            vmax[bb] = fmaxf(vmax[bb], vn);
          }
        }
      }
      if (t < T_STEPS) grid_barrier(bar, (unsigned)(NWG * (t + 1)));
    }
    if (wave == 0 && lane < 10) {
#pragma unroll
      for (int bb = 0; bb < 4; ++bb) out[bb * 10 + lane] = vmax[bb];
    }
  }
}

extern "C" void kernel_launch(void* const* d_in, const int* in_sizes, int n_in,
                              void* d_out, int out_size, void* d_ws, size_t ws_size,
                              hipStream_t stream) {
  const float* x  = (const float*)d_in[0];
  const float* u  = (const float*)d_in[1];
  const float* W1 = (const float*)d_in[2];
  const float* b1 = (const float*)d_in[3];
  const float* W2 = (const float*)d_in[4];
  const float* b2 = (const float*)d_in[5];
  const float* W3 = (const float*)d_in[6];
  const float* b3 = (const float*)d_in[7];
  char* ws = (char*)d_ws;
  if (ws_size < WS_NEEDED) return;                 // ~20.2 MB scratch required

  snn_init<<<(int)((Z1A_HALVES + 255) / 256), 256, 0, stream>>>(ws);
  pack_w<<<(int)((W1H_HALVES + 255) / 256), 256, 0, stream>>>(
      W1, (_Float16*)(ws + OFF_W1H), NK1, 3072, 1536, (long)W1H_HALVES);
  pack_w<<<(int)((W2H_HALVES + 255) / 256), 256, 0, stream>>>(
      W2, (_Float16*)(ws + OFF_W2H), NK2, 1536, 256, (long)W2H_HALVES);
  pack_w<<<(int)((W3H_HALVES + 255) / 256), 256, 0, stream>>>(
      W3, (_Float16*)(ws + OFF_W3H), NK3, 256, 10, (long)W3H_HALVES);
  encode_spikes<<<(int)((SPK_HALVES + 255) / 256), 256, 0, stream>>>(
      u, x, (_Float16*)(ws + OFF_SPK));
  snn_main<<<NWG, 256, 0, stream>>>(b1, b2, b3, (float*)d_out, ws);
}
